// MriViT_82308753260742
// MI455X (gfx1250) — compile-verified, matrix operand order checked by reading, not run on hardware
//
#include <hip/hip_runtime.h>
#include <hip/hip_bf16.h>
#include <math.h>

typedef _Float16 half_t;
typedef __attribute__((ext_vector_type(16))) _Float16 v16h;
typedef __attribute__((ext_vector_type(8)))  _Float16 v8h;
typedef __attribute__((ext_vector_type(8)))  float    v8f;

#define BATCH   64
#define SEQ     257
#define SP      288          // padded sequence (multiple of 32)
#define DMODEL  384
#define NHEAD   6
#define DHEAD   64
#define NBLK    8
#define FFDIM   1536
#define MROWS   (BATCH*SEQ)  // 16448 = 257*64
#define NPATCH  256
#define PATCHK  256
#define EMB_M   (BATCH*NPATCH) // 16384

// ---------------------------------------------------------------------------
// WMMA helpers (gfx1250, wave32): D(16x16,f32) = A(16x32,f16) * B(32x16,f16) + C
// A fragment: lane m(0..15) holds row m, K = {0..7, 16..23}; lane m+16 holds
// K = {8..15, 24..31}. B fragment mirrors with N as the lane axis, so a
// row-major (N x K) weight matrix loads contiguously (lane n reads row n).
// ---------------------------------------------------------------------------
__device__ __forceinline__ v16h load_frag(const half_t* base, int ld, int lane) {
  const int r  = lane & 15;
  const int ko = (lane >> 4) << 3;           // 0 or 8
  const half_t* p = base + (size_t)r * ld + ko;
  v8h lo = *(const v8h*)(p);                 // K = ko .. ko+7
  v8h hi = *(const v8h*)(p + 16);            // K = ko+16 .. ko+23
  return __builtin_shufflevector(lo, hi, 0,1,2,3,4,5,6,7,8,9,10,11,12,13,14,15);
}

__device__ __forceinline__ v8f wmma16(v16h a, v16h b, v8f c) {
  return __builtin_amdgcn_wmma_f32_16x16x32_f16(false, a, false, b, (short)0, c,
                                                false, false);
}

__device__ __forceinline__ v8f vzero8() {
  v8f z = {0.f,0.f,0.f,0.f,0.f,0.f,0.f,0.f};
  return z;
}

// ---------------------------------------------------------------------------
// fp32 -> fp16 weight conversion
// ---------------------------------------------------------------------------
__global__ void f32_to_f16_kernel(const float* __restrict__ a,
                                  half_t* __restrict__ o, size_t n) {
  size_t i = (size_t)blockIdx.x * blockDim.x + threadIdx.x;
  if (i < n) o[i] = (half_t)a[i];
}

// ---------------------------------------------------------------------------
// images (B,1,256,256) fp32 -> patches (B*256, 256) f16
// ---------------------------------------------------------------------------
__global__ void patchify_kernel(const float* __restrict__ img,
                                half_t* __restrict__ patches) {
  size_t idx = (size_t)blockIdx.x * blockDim.x + threadIdx.x;
  if (idx >= (size_t)EMB_M * PATCHK) return;
  int col = (int)(idx & 255);       // py*16+px
  int m   = (int)(idx >> 8);        // n*256 + patch
  int p   = m & 255;
  int n   = m >> 8;
  int ph = p >> 4, pw = p & 15;
  int py = col >> 4, px = col & 15;
  patches[idx] = (half_t)img[((size_t)n * 256 + (size_t)(ph*16 + py)) * 256 +
                             (size_t)(pw*16 + px)];
}

// ---------------------------------------------------------------------------
// cls token + sinusoidal positional embeddings into x (B,S,D) fp32
// ---------------------------------------------------------------------------
__global__ void poscls_kernel(float* __restrict__ x, const float* __restrict__ cls) {
  size_t idx = (size_t)blockIdx.x * blockDim.x + threadIdx.x;
  if (idx >= (size_t)BATCH * SEQ * DMODEL) return;
  int d = (int)(idx % DMODEL);
  int s = (int)((idx / DMODEL) % SEQ);
  int jeff = d & ~1;
  float ang = (float)s * powf(10000.0f, -(float)jeff / (float)DMODEL);
  float pe = (d & 1) ? cosf(ang) : sinf(ang);
  if (s == 0) x[idx] = cls[d] + pe;
  else        x[idx] += pe;
}

// ---------------------------------------------------------------------------
// LayerNorm: one wave per row (384 = 32 lanes * 12), output f16
// ---------------------------------------------------------------------------
__global__ void __launch_bounds__(256, 1)
layernorm_kernel(const float* __restrict__ x,
                 const float* __restrict__ g,
                 const float* __restrict__ bta,
                 half_t* __restrict__ out, int rows) {
  int wave = threadIdx.x >> 5;
  int lane = threadIdx.x & 31;
  int row  = blockIdx.x * (blockDim.x >> 5) + wave;
  if (row >= rows) return;
  const float* xr = x + (size_t)row * DMODEL;
  float v[12];
  float s = 0.f;
#pragma unroll
  for (int i = 0; i < 12; ++i) { v[i] = xr[lane + 32*i]; s += v[i]; }
#pragma unroll
  for (int o = 16; o >= 1; o >>= 1) s += __shfl_xor(s, o, 32);
  float mu = s * (1.0f / DMODEL);
  float q = 0.f;
#pragma unroll
  for (int i = 0; i < 12; ++i) { float d = v[i] - mu; q += d * d; }
#pragma unroll
  for (int o = 16; o >= 1; o >>= 1) q += __shfl_xor(q, o, 32);
  float rstd = rsqrtf(q * (1.0f / DMODEL) + 1e-5f);
  half_t* orow = out + (size_t)row * DMODEL;
#pragma unroll
  for (int i = 0; i < 12; ++i) {
    int c = lane + 32*i;
    orow[c] = (half_t)((v[i] - mu) * rstd * g[c] + bta[c]);
  }
}

// ---------------------------------------------------------------------------
// Generic WMMA GEMM: C(MxN) = act(A(MxK,f16) @ W(NxK,f16)^T + bias)
// One wave per 64x64 tile; __launch_bounds__(32,1) -> full VGPR file, no spill.
// mode: 0 = fp32 store (remap for patch-embed rows),
// 1 = GELU -> f16 store, 2 = fp32 residual accumulate.
// ---------------------------------------------------------------------------
__global__ void __launch_bounds__(32, 1)
gemm_wmma(const half_t* __restrict__ A,
          const half_t* __restrict__ W,
          const float* __restrict__ bias,
          float* __restrict__ Cf, half_t* __restrict__ Ch,
          int N, int K, int mode, int remap) {
  const int lane = threadIdx.x & 31;
  const int m0 = blockIdx.x * 64;
  const int n0 = blockIdx.y * 64;
  v8f acc[4][4];
#pragma unroll
  for (int i = 0; i < 4; ++i)
#pragma unroll
    for (int j = 0; j < 4; ++j) acc[i][j] = vzero8();

  // Hoist per-fragment row pointers (adds lane offsets once).
  const int r  = lane & 15;
  const int ko = (lane >> 4) << 3;
  const half_t* ap[4];
  const half_t* wp[4];
#pragma unroll
  for (int i = 0; i < 4; ++i) {
    ap[i] = A + ((size_t)(m0 + 16*i + r)) * K + ko;
    wp[i] = W + ((size_t)(n0 + 16*i + r)) * K + ko;
  }

  for (int k0 = 0; k0 < K; k0 += 32) {
    if (k0 + 32 < K) {   // gfx1250 global_prefetch of next K-chunk
      __builtin_prefetch(ap[0] + k0 + 32, 0, 1);
      __builtin_prefetch(wp[0] + k0 + 32, 0, 1);
    }
    v16h af[4];
#pragma unroll
    for (int i = 0; i < 4; ++i) {
      v8h lo = *(const v8h*)(ap[i] + k0);
      v8h hi = *(const v8h*)(ap[i] + k0 + 16);
      af[i] = __builtin_shufflevector(lo, hi, 0,1,2,3,4,5,6,7,8,9,10,11,12,13,14,15);
    }
#pragma unroll
    for (int j = 0; j < 4; ++j) {
      v8h lo = *(const v8h*)(wp[j] + k0);
      v8h hi = *(const v8h*)(wp[j] + k0 + 16);
      v16h bf = __builtin_shufflevector(lo, hi, 0,1,2,3,4,5,6,7,8,9,10,11,12,13,14,15);
#pragma unroll
      for (int i = 0; i < 4; ++i)
        acc[i][j] = wmma16(af[i], bf, acc[i][j]);
    }
  }

  const int rb = (lane >> 4) << 3;   // row base 0 or 8
  const int cl = lane & 15;          // column within 16
#pragma unroll
  for (int i = 0; i < 4; ++i)
#pragma unroll
    for (int j = 0; j < 4; ++j) {
      int colg = n0 + 16*j + cl;
      float bv = bias[colg];
#pragma unroll
      for (int r8 = 0; r8 < 8; ++r8) {
        int rowg = m0 + 16*i + rb + r8;
        float val = acc[i][j][r8] + bv;
        if (mode == 0) {
          int orow = remap ? ((rowg >> 8) * SEQ + 1 + (rowg & 255)) : rowg;
          Cf[(size_t)orow * N + colg] = val;
        } else if (mode == 1) {
          float gel = 0.5f * val * (1.0f + erff(val * 0.70710678118f));
          Ch[(size_t)rowg * N + colg] = (half_t)gel;
        } else {
          Cf[(size_t)rowg * N + colg] += val;
        }
      }
    }
}

// ---------------------------------------------------------------------------
// QKV projection: one wave per (b, h, 16-row tile of SP).
// Q,K -> (B,H,SP,DH) row-major f16; V -> transposed (B,H,DH,SP) f16.
// Padded rows (s >= 257) stored as zero.
// ---------------------------------------------------------------------------
__global__ void __launch_bounds__(32, 1)
qkv_kernel(const half_t* __restrict__ xln,
           const half_t* __restrict__ wq,
           const half_t* __restrict__ wk,
           const half_t* __restrict__ wv,
           const float* __restrict__ bq,
           const float* __restrict__ bk,
           const float* __restrict__ bv,
           half_t* __restrict__ qb, half_t* __restrict__ kb,
           half_t* __restrict__ vt) {
  const int lane = threadIdx.x & 31;
  const int t = blockIdx.x;   // 0..SP/16-1
  const int h = blockIdx.y;
  const int b = blockIdx.z;
  v8f aq[4], ak[4], av[4];
#pragma unroll
  for (int j = 0; j < 4; ++j) { aq[j] = vzero8(); ak[j] = vzero8(); av[j] = vzero8(); }

  const half_t* abase = xln + ((size_t)b * SEQ + (size_t)t * 16) * DMODEL + h * DHEAD;
  const half_t* wqh = wq + (size_t)h * DHEAD * DHEAD;
  const half_t* wkh = wk + (size_t)h * DHEAD * DHEAD;
  const half_t* wvh = wv + (size_t)h * DHEAD * DHEAD;
#pragma unroll
  for (int ks = 0; ks < 2; ++ks) {
    v16h af = load_frag(abase + 32*ks, DMODEL, lane);
#pragma unroll
    for (int j = 0; j < 4; ++j) {
      v16h bqf = load_frag(wqh + (size_t)(16*j) * DHEAD + 32*ks, DHEAD, lane);
      aq[j] = wmma16(af, bqf, aq[j]);
      v16h bkf = load_frag(wkh + (size_t)(16*j) * DHEAD + 32*ks, DHEAD, lane);
      ak[j] = wmma16(af, bkf, ak[j]);
      v16h bvf = load_frag(wvh + (size_t)(16*j) * DHEAD + 32*ks, DHEAD, lane);
      av[j] = wmma16(af, bvf, av[j]);
    }
  }
  const int rb = (lane >> 4) << 3;
  const int cl = lane & 15;
  const size_t bh = (size_t)b * NHEAD + h;
#pragma unroll
  for (int j = 0; j < 4; ++j) {
    int e = 16*j + cl;
    float bqv = bq[h*DHEAD + e];
    float bkv = bk[h*DHEAD + e];
    float bvv = bv[h*DHEAD + e];
#pragma unroll
    for (int r = 0; r < 8; ++r) {
      int s = t*16 + rb + r;
      bool valid = (s < SEQ);
      float qv = valid ? aq[j][r] + bqv : 0.f;
      float kv = valid ? ak[j][r] + bkv : 0.f;
      float vv = valid ? av[j][r] + bvv : 0.f;
      qb[(bh * SP + s) * DHEAD + e] = (half_t)qv;
      kb[(bh * SP + s) * DHEAD + e] = (half_t)kv;
      vt[(bh * DHEAD + e) * SP + s] = (half_t)vv;
    }
  }
}

// ---------------------------------------------------------------------------
// Attention: one wave per (b, h, 16-row Q tile). Scores via WMMA -> LDS,
// per-row softmax, P@V via WMMA (P fragment loaded from LDS), residual add.
// ---------------------------------------------------------------------------
__global__ void __launch_bounds__(32, 1)
attn_kernel(const half_t* __restrict__ qb,
            const half_t* __restrict__ kb,
            const half_t* __restrict__ vt,
            float* __restrict__ x) {
  __shared__ __align__(16) float  ps[16][SP];
  __shared__ __align__(16) half_t p16[16][SP];
  __shared__ float lsum[16];
  const int lane = threadIdx.x & 31;
  const int t = blockIdx.x;            // 0..16 (ceil(SEQ/16))
  const int h = blockIdx.y;
  const int b = blockIdx.z;
  const size_t bh = (size_t)b * NHEAD + h;
  const int rb = (lane >> 4) << 3;
  const int cl = lane & 15;

  v16h aq0 = load_frag(qb + (bh * SP + (size_t)t * 16) * DHEAD + 0,  DHEAD, lane);
  v16h aq1 = load_frag(qb + (bh * SP + (size_t)t * 16) * DHEAD + 32, DHEAD, lane);

  for (int jt = 0; jt < SP / 16; ++jt) {
    v8f sc = vzero8();
    sc = wmma16(aq0, load_frag(kb + (bh * SP + (size_t)jt*16) * DHEAD + 0,  DHEAD, lane), sc);
    sc = wmma16(aq1, load_frag(kb + (bh * SP + (size_t)jt*16) * DHEAD + 32, DHEAD, lane), sc);
#pragma unroll
    for (int r = 0; r < 8; ++r) ps[rb + r][jt*16 + cl] = sc[r] * 0.125f;
  }
  __syncthreads();

  if (lane < 16) {                      // per-row exact softmax (masked > SEQ)
    float m = -3.0e38f;
    for (int j = 0; j < SEQ; ++j) m = fmaxf(m, ps[lane][j]);
    float l = 0.f;
    for (int j = 0; j < SP; ++j) {
      float e = (j < SEQ) ? expf(ps[lane][j] - m) : 0.f;
      p16[lane][j] = (half_t)e;
      l += e;
    }
    lsum[lane] = l;
  }
  __syncthreads();

  v8f acco[4];
#pragma unroll
  for (int j = 0; j < 4; ++j) acco[j] = vzero8();
  for (int kt = 0; kt < SP / 32; ++kt) {
    v16h ap = load_frag(&p16[0][0] + kt*32, SP, lane);   // from LDS
#pragma unroll
    for (int j = 0; j < 4; ++j) {
      v16h bf = load_frag(vt + (bh * DHEAD + (size_t)(16*j)) * SP + kt*32, SP, lane);
      acco[j] = wmma16(ap, bf, acco[j]);
    }
  }
#pragma unroll
  for (int j = 0; j < 4; ++j) {
    int e = 16*j + cl;
#pragma unroll
    for (int r = 0; r < 8; ++r) {
      int s = t*16 + rb + r;
      if (s < SEQ) {
        float inv = 1.0f / lsum[rb + r];
        x[((size_t)b * SEQ + s) * DMODEL + h * DHEAD + e] += acco[j][r] * inv;
      }
    }
  }
}

// ---------------------------------------------------------------------------
// Classifier head on cls token + softmax(4)
// ---------------------------------------------------------------------------
__global__ void head_kernel(const float* __restrict__ x,
                            const float* __restrict__ wout,
                            const float* __restrict__ bout,
                            float* __restrict__ out) {
  int b = blockIdx.x * blockDim.x + threadIdx.x;
  if (b >= BATCH) return;
  const float* xr = x + (size_t)b * SEQ * DMODEL;
  float s[4];
  float m = -3.0e38f;
#pragma unroll
  for (int o = 0; o < 4; ++o) {
    float a = bout[o];
    for (int d = 0; d < DMODEL; ++d) a += xr[d] * wout[o*DMODEL + d];
    s[o] = a;
    m = fmaxf(m, a);
  }
  float l = 0.f;
#pragma unroll
  for (int o = 0; o < 4; ++o) { s[o] = expf(s[o] - m); l += s[o]; }
#pragma unroll
  for (int o = 0; o < 4; ++o) out[b*4 + o] = s[o] / l;
}

// ---------------------------------------------------------------------------
extern "C" void kernel_launch(void* const* d_in, const int* in_sizes, int n_in,
                              void* d_out, int out_size, void* d_ws, size_t ws_size,
                              hipStream_t stream) {
  (void)in_sizes; (void)n_in; (void)out_size; (void)ws_size;
  const float* images = (const float*)d_in[0];
  const float* Wmap   = (const float*)d_in[1];
  const float* bmap   = (const float*)d_in[2];
  const float* cls    = (const float*)d_in[3];
  const float* ln1g   = (const float*)d_in[4];
  const float* ln1b   = (const float*)d_in[5];
  const float* Wq     = (const float*)d_in[6];
  const float* bq     = (const float*)d_in[7];
  const float* Wk     = (const float*)d_in[8];
  const float* bk     = (const float*)d_in[9];
  const float* Wv     = (const float*)d_in[10];
  const float* bv     = (const float*)d_in[11];
  const float* ln2g   = (const float*)d_in[12];
  const float* ln2b   = (const float*)d_in[13];
  const float* W1     = (const float*)d_in[14];
  const float* b1     = (const float*)d_in[15];
  const float* W2     = (const float*)d_in[16];
  const float* b2     = (const float*)d_in[17];
  const float* Wout   = (const float*)d_in[18];
  const float* bout   = (const float*)d_in[19];
  float* out = (float*)d_out;

  // Workspace bump allocator
  char* ws = (char*)d_ws;
  size_t off = 0;
  auto alloc = [&](size_t bytes) -> void* {
    void* p = ws + off;
    off = (off + bytes + 255) & ~(size_t)255;
    return p;
  };
  float*  x       = (float*) alloc((size_t)MROWS * DMODEL * 4);
  half_t* xln     = (half_t*)alloc((size_t)(MROWS + 64) * DMODEL * 2); // +64 OOB pad
  half_t* patches = (half_t*)alloc((size_t)EMB_M * PATCHK * 2);
  half_t* hbuf    = (half_t*)alloc((size_t)MROWS * FFDIM * 2);
  half_t* qbuf    = (half_t*)alloc((size_t)BATCH * NHEAD * SP * DHEAD * 2);
  half_t* kbuf    = (half_t*)alloc((size_t)BATCH * NHEAD * SP * DHEAD * 2);
  half_t* vtbuf   = (half_t*)alloc((size_t)BATCH * NHEAD * DHEAD * SP * 2);
  half_t* wmap16  = (half_t*)alloc((size_t)DMODEL * PATCHK * 2);
  half_t* wq16    = (half_t*)alloc((size_t)NBLK * NHEAD * DHEAD * DHEAD * 2);
  half_t* wk16    = (half_t*)alloc((size_t)NBLK * NHEAD * DHEAD * DHEAD * 2);
  half_t* wv16    = (half_t*)alloc((size_t)NBLK * NHEAD * DHEAD * DHEAD * 2);
  half_t* w116    = (half_t*)alloc((size_t)NBLK * FFDIM * DMODEL * 2);
  half_t* w216    = (half_t*)alloc((size_t)NBLK * DMODEL * FFDIM * 2);

  auto cvt = [&](const float* src, half_t* dst, size_t n) {
    f32_to_f16_kernel<<<(unsigned)((n + 255) / 256), 256, 0, stream>>>(src, dst, n);
  };
  cvt(Wmap, wmap16, (size_t)DMODEL * PATCHK);
  cvt(Wq,   wq16,   (size_t)NBLK * NHEAD * DHEAD * DHEAD);
  cvt(Wk,   wk16,   (size_t)NBLK * NHEAD * DHEAD * DHEAD);
  cvt(Wv,   wv16,   (size_t)NBLK * NHEAD * DHEAD * DHEAD);
  cvt(W1,   w116,   (size_t)NBLK * FFDIM * DMODEL);
  cvt(W2,   w216,   (size_t)NBLK * DMODEL * FFDIM);

  // Patch embedding
  patchify_kernel<<<(unsigned)(((size_t)EMB_M * PATCHK + 255) / 256), 256, 0, stream>>>(
      images, patches);
  gemm_wmma<<<dim3(EMB_M / 64, DMODEL / 64), 32, 0, stream>>>(
      patches, wmap16, bmap, x, nullptr, DMODEL, PATCHK, /*mode=*/0, /*remap=*/1);
  poscls_kernel<<<(unsigned)(((size_t)BATCH * SEQ * DMODEL + 255) / 256), 256, 0, stream>>>(
      x, cls);

  const unsigned lnGrid = (MROWS + 7) / 8;
  for (int blk = 0; blk < NBLK; ++blk) {
    layernorm_kernel<<<lnGrid, 256, 0, stream>>>(
        x, ln1g + blk * DMODEL, ln1b + blk * DMODEL, xln, MROWS);
    qkv_kernel<<<dim3(SP / 16, NHEAD, BATCH), 32, 0, stream>>>(
        xln,
        wq16 + (size_t)blk * NHEAD * DHEAD * DHEAD,
        wk16 + (size_t)blk * NHEAD * DHEAD * DHEAD,
        wv16 + (size_t)blk * NHEAD * DHEAD * DHEAD,
        bq + (size_t)blk * NHEAD * DHEAD,
        bk + (size_t)blk * NHEAD * DHEAD,
        bv + (size_t)blk * NHEAD * DHEAD,
        qbuf, kbuf, vtbuf);
    attn_kernel<<<dim3((SEQ + 15) / 16, NHEAD, BATCH), 32, 0, stream>>>(
        qbuf, kbuf, vtbuf, x);
    layernorm_kernel<<<lnGrid, 256, 0, stream>>>(
        x, ln2g + blk * DMODEL, ln2b + blk * DMODEL, xln, MROWS);
    gemm_wmma<<<dim3(MROWS / 64, FFDIM / 64), 32, 0, stream>>>(
        xln, w116 + (size_t)blk * FFDIM * DMODEL, b1 + (size_t)blk * FFDIM,
        nullptr, hbuf, FFDIM, DMODEL, /*mode=*/1, /*remap=*/0);
    gemm_wmma<<<dim3(MROWS / 64, DMODEL / 64), 32, 0, stream>>>(
        hbuf, w216 + (size_t)blk * DMODEL * FFDIM, b2 + (size_t)blk * DMODEL,
        x, nullptr, DMODEL, FFDIM, /*mode=*/2, /*remap=*/0);
  }

  head_kernel<<<1, 64, 0, stream>>>(x, Wout, bout, out);
}
